// TransformerBlock_46076409151569
// MI455X (gfx1250) — matrix operand order repaired and checked by reading, not compiled, not run on hardware
//
#include <hip/hip_runtime.h>

typedef __attribute__((ext_vector_type(16))) __bf16 v16bf;
typedef __attribute__((ext_vector_type(8)))  float  v8f;
typedef int v4i __attribute__((vector_size(4 * sizeof(int))));
typedef unsigned int   u32;
typedef unsigned short u16;

union Frag { v16bf v; u32 u[8]; };

typedef __attribute__((address_space(1))) v4i gv4i;   // global int4
typedef __attribute__((address_space(3))) v4i lv4i;   // LDS int4

#if defined(__has_builtin)
#if __has_builtin(__builtin_amdgcn_global_load_async_to_lds_b128)
#define ASYNC_LDS 1
#endif
#if __has_builtin(__builtin_amdgcn_s_wait_asynccnt)
#define HAVE_WAIT_ASYNC 1
#endif
#endif

__device__ __forceinline__ void async_copy16(const void* g, void* l) {
#if defined(ASYNC_LDS)
  __builtin_amdgcn_global_load_async_to_lds_b128(
      (gv4i*)(uintptr_t)g, (lv4i*)(u32)(uintptr_t)l, 0, 0);
#else
  *(v4i*)l = *(const v4i*)g;
#endif
}

__device__ __forceinline__ void wait_async0() {
#if defined(ASYNC_LDS)
#if defined(HAVE_WAIT_ASYNC)
  __builtin_amdgcn_s_wait_asynccnt(0);
#else
  asm volatile("s_wait_asynccnt 0" ::: "memory");
#endif
#endif
}

__device__ __forceinline__ u16 f2bf(float f) {
  u32 u = __float_as_uint(f);
  u = (u + 0x7FFFu + ((u >> 16) & 1u)) >> 16;   // round-to-nearest-even
  return (u16)u;
}

// ---------------------------------------------------------------------------
// fp32 -> bf16 elementwise convert
// ---------------------------------------------------------------------------
__global__ __launch_bounds__(256) void cvt_bf16(const float* __restrict__ in,
                                                u16* __restrict__ out, int n) {
  int i = blockIdx.x * 256 + threadIdx.x;
  if (i < n) out[i] = f2bf(in[i]);
}

// ---------------------------------------------------------------------------
// LayerNorm: one block per row, bf16 output
// ---------------------------------------------------------------------------
__global__ __launch_bounds__(256) void ln_bf16(const float* __restrict__ x,
                                               const float* __restrict__ g,
                                               const float* __restrict__ bta,
                                               u16* __restrict__ out, int D) {
  int row = blockIdx.x;
  const float* xr = x + (size_t)row * D;
  float s = 0.f, s2 = 0.f;
  for (int i = threadIdx.x; i < D; i += 256) { float v = xr[i]; s += v; s2 += v * v; }
  __shared__ float rs[256], rs2[256];
  rs[threadIdx.x] = s; rs2[threadIdx.x] = s2;
  __syncthreads();
  for (int off = 128; off > 0; off >>= 1) {
    if (threadIdx.x < (unsigned)off) {
      rs[threadIdx.x]  += rs[threadIdx.x + off];
      rs2[threadIdx.x] += rs2[threadIdx.x + off];
    }
    __syncthreads();
  }
  float mean = rs[0] / (float)D;
  float var  = rs2[0] / (float)D - mean * mean;
  float rstd = rsqrtf(var + 1e-5f);
  for (int i = threadIdx.x; i < D; i += 256)
    out[(size_t)row * D + i] = f2bf((xr[i] - mean) * rstd * g[i] + bta[i]);
}

// ---------------------------------------------------------------------------
// bf16 WMMA GEMM: C[M,N] = act(A[M,K] @ B[K,N] + bias (+ resid))
// Block tile 128x128, BK=32. 8 waves: 4 along M (32 rows) x 2 along N (64 cols)
// -> each wave 2x4 fragments = 8 wmma per k-step.
// Double-buffered LDS; A tile staged with async global->LDS b128 (ASYNCcnt),
// B tile prefetched with wide b128 loads into registers then stored transposed
// ([n][k]) so each lane's fragment pair read is one aligned dword.
// A frag (16x32): lane m = lane&15, vgpr g at k = 16*half + 2g.
// B frag (32x16): lane n = lane&15, vgpr g at k = 16*half + 2g.
// ---------------------------------------------------------------------------
template <int RELU, int RESID, int WF, int WB>
__global__ __launch_bounds__(256) void gemm_bf16_wmma(
    const u16* __restrict__ A, const u16* __restrict__ B,
    const float* __restrict__ bias, const float* __restrict__ resid,
    float* __restrict__ outF, u16* __restrict__ outB,
    int M, int N, int K) {
  __shared__ __align__(16) u16 lA[2][128][40];
  __shared__ __align__(16) u16 lB[2][128][40];

  const int tid  = threadIdx.x;
  const int lane = tid & 31;
  const int l16  = lane & 15;
  const int hi16 = lane >> 4;
  const int wid  = tid >> 5;
  const int wm   = wid & 3;     // 4 waves along M (32 rows each)
  const int wn   = wid >> 2;    // 2 waves along N (64 cols each)

  const int m0 = blockIdx.y * 128;
  const int n0 = blockIdx.x * 128;

  // A-stage: 128x32 bf16 = 512 x 16B chunks; 2 chunks per thread
  const int arow0 = tid >> 2,          acc0 = (tid & 3) * 8;
  const int arow1 = (tid + 256) >> 2,  acc1 = ((tid + 256) & 3) * 8;
  // B-stage: 32x128 bf16 = 512 x 16B chunks; 2 chunks per thread
  const int brow0 = tid >> 4,          bcc0 = (tid & 15) * 8;
  const int brow1 = (tid + 256) >> 4,  bcc1 = ((tid + 256) & 15) * 8;

  v8f zf = {0.f, 0.f, 0.f, 0.f, 0.f, 0.f, 0.f, 0.f};
  v8f acc[2][4];
#pragma unroll
  for (int i = 0; i < 2; ++i)
#pragma unroll
    for (int j = 0; j < 4; ++j) acc[i][j] = zf;

  v4i breg[2];

  // ---- prologue: stage k0 = 0 into buffer 0 ----
  async_copy16(&A[(size_t)(m0 + arow0) * K + acc0], &lA[0][arow0][acc0]);
  async_copy16(&A[(size_t)(m0 + arow1) * K + acc1], &lA[0][arow1][acc1]);
  breg[0] = *(const v4i*)(&B[(size_t)brow0 * N + n0 + bcc0]);
  breg[1] = *(const v4i*)(&B[(size_t)brow1 * N + n0 + bcc1]);
#pragma unroll
  for (int jj = 0; jj < 4; ++jj) {
    u32 w0 = (u32)breg[0][jj], w1 = (u32)breg[1][jj];
    lB[0][bcc0 + 2 * jj + 0][brow0] = (u16)(w0 & 0xFFFFu);
    lB[0][bcc0 + 2 * jj + 1][brow0] = (u16)(w0 >> 16);
    lB[0][bcc1 + 2 * jj + 0][brow1] = (u16)(w1 & 0xFFFFu);
    lB[0][bcc1 + 2 * jj + 1][brow1] = (u16)(w1 >> 16);
  }
  wait_async0();
  __syncthreads();

  int buf = 0;
  for (int k0 = 0; k0 < K; k0 += 32) {
    const int nxt = buf ^ 1;
    const bool more = (k0 + 32) < K;
    if (more) {
      // prefetch next stage: A via async-LDS DMA, B into registers (b128)
      async_copy16(&A[(size_t)(m0 + arow0) * K + k0 + 32 + acc0], &lA[nxt][arow0][acc0]);
      async_copy16(&A[(size_t)(m0 + arow1) * K + k0 + 32 + acc1], &lA[nxt][arow1][acc1]);
      breg[0] = *(const v4i*)(&B[(size_t)(k0 + 32 + brow0) * N + n0 + bcc0]);
      breg[1] = *(const v4i*)(&B[(size_t)(k0 + 32 + brow1) * N + n0 + bcc1]);
    }

    Frag a[2], b[4];
#pragma unroll
    for (int i = 0; i < 2; ++i) {
      int row = wm * 32 + i * 16 + l16;
#pragma unroll
      for (int g = 0; g < 8; ++g) {
        int kb = 16 * hi16 + 2 * g;
        a[i].u[g] = *(const u32*)(&lA[buf][row][kb]);
      }
    }
#pragma unroll
    for (int j = 0; j < 4; ++j) {
      int n = wn * 64 + j * 16 + l16;
#pragma unroll
      for (int g = 0; g < 8; ++g)
        b[j].u[g] = *(const u32*)(&lB[buf][n][16 * hi16 + 2 * g]);
    }
#pragma unroll
    for (int i = 0; i < 2; ++i)
#pragma unroll
      for (int j = 0; j < 4; ++j)
        acc[i][j] = __builtin_amdgcn_wmma_f32_16x16x32_bf16(
            false, a[i].v, false, b[j].v, (short)0, acc[i][j], false, false);

    if (more) {
#pragma unroll
      for (int jj = 0; jj < 4; ++jj) {
        u32 w0 = (u32)breg[0][jj], w1 = (u32)breg[1][jj];
        lB[nxt][bcc0 + 2 * jj + 0][brow0] = (u16)(w0 & 0xFFFFu);
        lB[nxt][bcc0 + 2 * jj + 1][brow0] = (u16)(w0 >> 16);
        lB[nxt][bcc1 + 2 * jj + 0][brow1] = (u16)(w1 & 0xFFFFu);
        lB[nxt][bcc1 + 2 * jj + 1][brow1] = (u16)(w1 >> 16);
      }
      wait_async0();
      __syncthreads();
    }
    buf = nxt;
  }

  // epilogue: D layout (ISA): vgpr e -> M = e + 8*hi16, lane -> N = l16
#pragma unroll
  for (int i = 0; i < 2; ++i)
#pragma unroll
    for (int j = 0; j < 4; ++j) {
      int col = n0 + wn * 64 + j * 16 + l16;
      float bv = bias[col];
#pragma unroll
      for (int e = 0; e < 8; ++e) {
        int row = m0 + wm * 32 + i * 16 + e + 8 * hi16;
        float v = acc[i][j][e] + bv;
        if (RESID) v += resid[(size_t)row * N + col];
        if (RELU)  v = fmaxf(v, 0.f);
        if (WF)    outF[(size_t)row * N + col] = v;
        if (WB)    outB[(size_t)row * N + col] = f2bf(v);
      }
    }
}

// ---------------------------------------------------------------------------
// Flash attention (causal), per (64-query tile, head, batch). dk = 64.
// Q,K,V,O are bf16 [B*S, D] with head h occupying columns h*64..h*64+63.
// Q and K tiles staged via async global->LDS b128 DMA; V staged through
// registers (2x b128 loads) then stored transposed.
// ---------------------------------------------------------------------------
__global__ __launch_bounds__(256) void attn_wmma(
    const u16* __restrict__ Q, const u16* __restrict__ Km,
    const u16* __restrict__ Vm, u16* __restrict__ O, int S, int D) {
  const int qt = blockIdx.x;
  const int h  = blockIdx.y;
  const int b  = blockIdx.z;
  const float scale = 0.125f;  // 1/sqrt(64)

  __shared__ __align__(16) u16 lQ[64][72];
  __shared__ __align__(16) u16 lK[64][72];    // [key][dk]
  __shared__ __align__(16) u16 lVT[64][72];   // [dk][key]
  __shared__ __align__(16) u16 lP[64][72];    // probs bf16
  __shared__ float lS[64][68];
  __shared__ float rowM[64], rowL[64], rowA[64];

  const int tid  = threadIdx.x;
  const int lane = tid & 31;
  const int l16  = lane & 15;
  const int hi16 = lane >> 4;
  const int wid  = tid >> 5;
  const int wr   = wid >> 1;   // 4 waves along rows (16 each)
  const int wc   = wid & 1;    // 2 waves along cols (32 each)

  const size_t base = ((size_t)b * S) * D + (size_t)h * 64;

  // tile staging geometry: 64x64 bf16 = 512 x 16B chunks; 2 per thread
  const int sr0 = tid >> 3,          scc0 = (tid & 7) * 8;
  const int sr1 = (tid + 256) >> 3,  scc1 = ((tid + 256) & 7) * 8;

  // Q tile via async DMA (completion folded into first iteration's wait)
  async_copy16(&Q[base + (size_t)(qt * 64 + sr0) * D + scc0], &lQ[sr0][scc0]);
  async_copy16(&Q[base + (size_t)(qt * 64 + sr1) * D + scc1], &lQ[sr1][scc1]);
  if (tid < 64) { rowM[tid] = -1e30f; rowL[tid] = 0.f; }

  v8f zf = {0.f, 0.f, 0.f, 0.f, 0.f, 0.f, 0.f, 0.f};
  v8f o[2]; o[0] = zf; o[1] = zf;

  for (int t = 0; t <= qt; ++t) {
    // K tile via async DMA; V tile via wide register loads + transposed store
    async_copy16(&Km[base + (size_t)(t * 64 + sr0) * D + scc0], &lK[sr0][scc0]);
    async_copy16(&Km[base + (size_t)(t * 64 + sr1) * D + scc1], &lK[sr1][scc1]);
    v4i vreg[2];
    vreg[0] = *(const v4i*)(&Vm[base + (size_t)(t * 64 + sr0) * D + scc0]);
    vreg[1] = *(const v4i*)(&Vm[base + (size_t)(t * 64 + sr1) * D + scc1]);
#pragma unroll
    for (int jj = 0; jj < 4; ++jj) {
      u32 w0 = (u32)vreg[0][jj], w1 = (u32)vreg[1][jj];
      lVT[scc0 + 2 * jj + 0][sr0] = (u16)(w0 & 0xFFFFu);
      lVT[scc0 + 2 * jj + 1][sr0] = (u16)(w0 >> 16);
      lVT[scc1 + 2 * jj + 0][sr1] = (u16)(w1 & 0xFFFFu);
      lVT[scc1 + 2 * jj + 1][sr1] = (u16)(w1 >> 16);
    }
    wait_async0();
    __syncthreads();

    // S = Q @ K^T
    v8f sc[2]; sc[0] = zf; sc[1] = zf;
    const int qrow = wr * 16 + l16;
#pragma unroll
    for (int ks = 0; ks < 2; ++ks) {
      Frag a;
#pragma unroll
      for (int g = 0; g < 8; ++g) {
        int kb = 32 * ks + 16 * hi16 + 2 * g;
        a.u[g] = *(const u32*)(&lQ[qrow][kb]);
      }
#pragma unroll
      for (int j = 0; j < 2; ++j) {
        Frag bfr;
        int n = wc * 32 + j * 16 + l16;
#pragma unroll
        for (int g = 0; g < 8; ++g)
          bfr.u[g] = *(const u32*)(&lK[n][32 * ks + 16 * hi16 + 2 * g]);
        sc[j] = __builtin_amdgcn_wmma_f32_16x16x32_bf16(
            false, a.v, false, bfr.v, (short)0, sc[j], false, false);
      }
    }
#pragma unroll
    for (int j = 0; j < 2; ++j)
#pragma unroll
      for (int e = 0; e < 8; ++e) {
        int r = wr * 16 + e + 8 * hi16;
        int c = wc * 32 + j * 16 + l16;
        int gq = qt * 64 + r, gk = t * 64 + c;
        lS[r][c] = (gk <= gq) ? sc[j][e] * scale : -1e30f;
      }
    __syncthreads();

    // online softmax per row
    if (tid < 64) {
      int r = tid;
      float tmax = -1e30f;
      for (int c = 0; c < 64; ++c) tmax = fmaxf(tmax, lS[r][c]);
      float nm = fmaxf(rowM[r], tmax);
      float al = __expf(rowM[r] - nm);
      float s = 0.f;
      for (int c = 0; c < 64; ++c) {
        float pv = __expf(lS[r][c] - nm);
        lP[r][c] = f2bf(pv);
        s += pv;
      }
      rowL[r] = rowL[r] * al + s;
      rowA[r] = al;
      rowM[r] = nm;
    }
    __syncthreads();

    // rescale O, then O += P @ V
#pragma unroll
    for (int j = 0; j < 2; ++j)
#pragma unroll
      for (int e = 0; e < 8; ++e)
        o[j][e] *= rowA[wr * 16 + e + 8 * hi16];
#pragma unroll
    for (int ks = 0; ks < 2; ++ks) {
      Frag a;
#pragma unroll
      for (int g = 0; g < 8; ++g) {
        int kb = 32 * ks + 16 * hi16 + 2 * g;
        a.u[g] = *(const u32*)(&lP[qrow][kb]);
      }
#pragma unroll
      for (int j = 0; j < 2; ++j) {
        Frag bfr;
        int n = wc * 32 + j * 16 + l16;
#pragma unroll
        for (int g = 0; g < 8; ++g)
          bfr.u[g] = *(const u32*)(&lVT[n][32 * ks + 16 * hi16 + 2 * g]);
        o[j] = __builtin_amdgcn_wmma_f32_16x16x32_bf16(
            false, a.v, false, bfr.v, (short)0, o[j], false, false);
      }
    }
    __syncthreads();
  }

#pragma unroll
  for (int j = 0; j < 2; ++j)
#pragma unroll
    for (int e = 0; e < 8; ++e) {
      int r = wr * 16 + e + 8 * hi16;
      int c = wc * 32 + j * 16 + l16;
      float v = o[j][e] / rowL[r];
      O[base + (size_t)(qt * 64 + r) * D + c] = f2bf(v);
    }
}

// ---------------------------------------------------------------------------
// host-side orchestration
// ---------------------------------------------------------------------------
extern "C" void kernel_launch(void* const* d_in, const int* in_sizes, int n_in,
                              void* d_out, int out_size, void* d_ws, size_t ws_size,
                              hipStream_t stream) {
  (void)in_sizes; (void)n_in; (void)out_size; (void)ws_size;
  const int Bb = 2, S = 2048, D = 1024, H = 16, FF = 4096;
  const int M = Bb * S;

  const float* x   = (const float*)d_in[0];
  /* d_in[1] = mask (bool) — causal mask computed analytically */
  const float* Wq  = (const float*)d_in[2];
  const float* bq  = (const float*)d_in[3];
  const float* Wk  = (const float*)d_in[4];
  const float* bk  = (const float*)d_in[5];
  const float* Wv  = (const float*)d_in[6];
  const float* bv  = (const float*)d_in[7];
  const float* Wo  = (const float*)d_in[8];
  const float* bo  = (const float*)d_in[9];
  const float* W1  = (const float*)d_in[10];
  const float* b1  = (const float*)d_in[11];
  const float* W2  = (const float*)d_in[12];
  const float* b2  = (const float*)d_in[13];
  const float* g1  = (const float*)d_in[14];
  const float* be1 = (const float*)d_in[15];
  const float* g2  = (const float*)d_in[16];
  const float* be2 = (const float*)d_in[17];

  char* p = (char*)d_ws;
  auto take = [&](size_t bytes) -> char* {
    char* r = p; p += (bytes + 255) & ~(size_t)255; return r;
  };
  u16*   wqB = (u16*)take((size_t)D * D * 2);
  u16*   wkB = (u16*)take((size_t)D * D * 2);
  u16*   wvB = (u16*)take((size_t)D * D * 2);
  u16*   woB = (u16*)take((size_t)D * D * 2);
  u16*   w1B = (u16*)take((size_t)D * FF * 2);
  u16*   w2B = (u16*)take((size_t)FF * D * 2);
  u16*   xnB = (u16*)take((size_t)M * D * 2);
  u16*   qB  = (u16*)take((size_t)M * D * 2);
  u16*   kB  = (u16*)take((size_t)M * D * 2);
  u16*   vB  = (u16*)take((size_t)M * D * 2);
  u16*   aB  = (u16*)take((size_t)M * D * 2);
  float* hF  = (float*)take((size_t)M * D * 4);
  u16*   hnB = (u16*)take((size_t)M * D * 2);
  u16*   f1B = (u16*)take((size_t)M * FF * 2);

  const int nDD = D * D, nDF = D * FF;
  cvt_bf16<<<(nDD + 255) / 256, 256, 0, stream>>>(Wq, wqB, nDD);
  cvt_bf16<<<(nDD + 255) / 256, 256, 0, stream>>>(Wk, wkB, nDD);
  cvt_bf16<<<(nDD + 255) / 256, 256, 0, stream>>>(Wv, wvB, nDD);
  cvt_bf16<<<(nDD + 255) / 256, 256, 0, stream>>>(Wo, woB, nDD);
  cvt_bf16<<<(nDF + 255) / 256, 256, 0, stream>>>(W1, w1B, nDF);
  cvt_bf16<<<(nDF + 255) / 256, 256, 0, stream>>>(W2, w2B, nDF);

  // xn = LN(x)
  ln_bf16<<<M, 256, 0, stream>>>(x, g1, be1, xnB, D);

  // q,k,v projections (bf16 outputs for attention)
  dim3 gP(D / 128, M / 128);
  gemm_bf16_wmma<0,0,0,1><<<gP, 256, 0, stream>>>(xnB, wqB, bq, nullptr, nullptr, qB, M, D, D);
  gemm_bf16_wmma<0,0,0,1><<<gP, 256, 0, stream>>>(xnB, wkB, bk, nullptr, nullptr, kB, M, D, D);
  gemm_bf16_wmma<0,0,0,1><<<gP, 256, 0, stream>>>(xnB, wvB, bv, nullptr, nullptr, vB, M, D, D);

  // flash attention (causal)
  attn_wmma<<<dim3(S / 64, H, Bb), 256, 0, stream>>>(qB, kB, vB, aB, S, D);

  // h = x + attn @ Wo + bo
  gemm_bf16_wmma<0,1,1,0><<<gP, 256, 0, stream>>>(aB, woB, bo, x, hF, nullptr, M, D, D);

  // hn = LN(h)
  ln_bf16<<<M, 256, 0, stream>>>(hF, g2, be2, hnB, D);

  // f1 = relu(hn @ W1 + b1)
  gemm_bf16_wmma<1,0,0,1><<<dim3(FF / 128, M / 128), 256, 0, stream>>>(
      hnB, w1B, b1, nullptr, nullptr, f1B, M, FF, D);

  // out = h + f1 @ W2 + b2
  gemm_bf16_wmma<0,1,1,0><<<dim3(D / 128, M / 128), 256, 0, stream>>>(
      f1B, w2B, b2, hF, (float*)d_out, nullptr, M, D, FF);
}
